// HardAttn_85882166051070
// MI455X (gfx1250) — matrix operand order repaired and checked by reading, not compiled, not run on hardware
//
#include <hip/hip_runtime.h>

typedef _Float16 f16;
typedef __attribute__((ext_vector_type(16))) _Float16 v16h;
typedef __attribute__((ext_vector_type(8)))  _Float16 v8h;
typedef __attribute__((ext_vector_type(8)))  float    v8f;

#define B_      4
#define DIM_    1024
#define L_      4096
#define NCODES_ 512
#define CDIM_   256
#define H_      8
#define D_      128
#define LTILE   128
#define KROW    136   // f16 elems per padded K/Q row (128 + 8 pad = 272 B)
#define OROW    129   // float elems per padded out row (conflict-free transpose)
#define KHALF   256   // K rows staged per phase

// Dynamic LDS layout (bytes): 105 KB -> 3 workgroups per WGP (320 KB LDS)
#define K_BYTES    (KHALF * KROW * 2)       // 69632: staged K half (f16)
#define Q_OFF      K_BYTES
#define Q_BYTES    (LTILE * KROW * 2)       // 34816: staged Q tile (f16)
#define IDX_OFF    (Q_OFF + Q_BYTES)        // 104448
#define SMEM_BYTES (IDX_OFF + LTILE * 4)    // 104960
// out-gather buffer (128*129*4 = 66048 B) aliases the K region after argmax.

// ---------------------------------------------------------------------------
// Kernel 1: k = codes @ Wk (-> f16), v = codes @ Wv (-> f32), layout [H][N][D]
// ---------------------------------------------------------------------------
__global__ void __launch_bounds__(256)
build_kv_kernel(const float* __restrict__ codes,
                const float* __restrict__ Wk,
                const float* __restrict__ Wv,
                f16* __restrict__ kf, float* __restrict__ vf) {
  __shared__ float crow[CDIM_];
  const int n = blockIdx.x;
  const int t = threadIdx.x;
  crow[t] = codes[n * CDIM_ + t];
  __syncthreads();
  float ak[4] = {0.f, 0.f, 0.f, 0.f};
  float av[4] = {0.f, 0.f, 0.f, 0.f};
  for (int c = 0; c < CDIM_; ++c) {
    const float cv = crow[c];
    const float* wk = Wk + (size_t)c * (H_ * D_);
    const float* wv = Wv + (size_t)c * (H_ * D_);
#pragma unroll
    for (int p = 0; p < 4; ++p) {
      const int j = t + p * 256;
      ak[p] = fmaf(cv, wk[j], ak[p]);
      av[p] = fmaf(cv, wv[j], av[p]);
    }
  }
#pragma unroll
  for (int p = 0; p < 4; ++p) {
    const int j = t + p * 256;
    const int hh = j >> 7, d = j & 127;
    const size_t o = ((size_t)hh * NCODES_ + n) * D_ + d;
    kf[o] = (f16)ak[p];
    vf[o] = av[p];
  }
}

// ---------------------------------------------------------------------------
// Kernel 2: per (b,h,l-tile): logits via v_wmma_f32_16x16x32_f16 (2-way tile
// interleave), running argmax, async-LDS K staging in 2 phases, gather +
// coalesced transposed write-out.
// ---------------------------------------------------------------------------
__global__ void __launch_bounds__(256)
hard_attn_kernel(const float* __restrict__ x,
                 const f16* __restrict__ kf,
                 const float* __restrict__ vf,
                 float* __restrict__ out,
                 float* __restrict__ idxout) {
  extern __shared__ char smem[];
  f16*   k_lds   = (f16*)smem;
  f16*   q_lds   = (f16*)(smem + Q_OFF);
  int*   idx_lds = (int*)(smem + IDX_OFF);
  float* o_lds   = (float*)smem;   // aliases K region; used only after barrier

  const int t    = threadIdx.x;
  const int lane = t & 31;
  const int wid  = t >> 5;         // wave 0..7, owns l-rows [wid*16, wid*16+16)
  const int half = lane >> 4;      // 0 or 1 (16-lane half)
  const int lm   = lane & 15;
  const int l0   = blockIdx.x * LTILE;
  const int h    = blockIdx.y;
  const int b    = blockIdx.z;

  // ---- async stage of one 256-row K half: ASYNCcnt-tracked, no VGPR trip ---
  const unsigned ldsb = (unsigned)(uintptr_t)(void*)k_lds;  // LDS byte offset
  const unsigned long long kgbase =
      (unsigned long long)(uintptr_t)(kf + (size_t)h * NCODES_ * D_);
  auto stageK = [&](int p) {
    const unsigned long long gb = kgbase + (unsigned long long)p * (KHALF * D_ * 2);
    for (int it = 0; it < (KHALF * D_ * 2) / 16 / 256; ++it) {  // 16 iters
      const int g = it * 256 + t;           // 16-byte chunk id
      const int n = g >> 4, c = g & 15;     // packed row 256B = 16 chunks
      const unsigned la = ldsb + (unsigned)(n * (KROW * 2) + c * 16);
      const unsigned go = (unsigned)g * 16u;
      asm volatile("global_load_async_to_lds_b128 %0, %1, %2 offset:0"
                   :: "v"(la), "v"(go), "s"(gb) : "memory");
    }
  };

  stageK(0);

  // ---- stage Q tile: x[b, h*128+d, l0+ll] (coalesced along l) -> f16 ----
  {
    const float* xb = x + ((size_t)b * DIM_ + (size_t)h * D_) * L_ + l0;
    for (int e = t; e < D_ * LTILE; e += 256) {
      const int d = e >> 7, ll = e & 127;
      q_lds[ll * KROW + d] = (f16)xb[(size_t)d * L_ + ll];
    }
  }
  asm volatile("s_wait_asynccnt 0" ::: "memory");
  __syncthreads();

  // ---- resident A fragments: 16 query rows x K=128 (4 k-steps of 32) ----
  // 16-bit A layout: half 0 -> K [k0..k0+7 | k0+16..k0+23], half 1 -> +8.
  v16h a[4];
  {
    const f16* ab = q_lds + (wid * 16 + lm) * KROW + half * 8;
#pragma unroll
    for (int ks = 0; ks < 4; ++ks) {
      v8h lo = *(const v8h*)(ab + ks * 32);
      v8h hi = *(const v8h*)(ab + ks * 32 + 16);
      a[ks] = __builtin_shufflevector(lo, hi, 0, 1, 2, 3, 4, 5, 6, 7,
                                      8, 9, 10, 11, 12, 13, 14, 15);
    }
  }

  float mv[8];
  int   mi[8];
#pragma unroll
  for (int j = 0; j < 8; ++j) { mv[j] = -3.402823466e38f; mi[j] = 0; }

  // ---- two phases of 256 codes; 2-way interleaved 16-code tiles ----
  for (int p = 0; p < 2; ++p) {
    if (p == 1) {
      __syncthreads();                       // all waves done with half 0
      stageK(1);
      asm volatile("s_wait_asynccnt 0" ::: "memory");
      __syncthreads();
    }
    for (int nt = 0; nt < KHALF / 16; nt += 2) {
      v8f c0 = {};
      v8f c1 = {};
      // 16-bit B layout: lane = column n, half 0 -> K [0..7 | 16..23], half 1 -> +8
      const f16* b0 = k_lds + (nt * 16 + lm) * KROW + half * 8;
      const f16* b1 = b0 + 16 * KROW;
#pragma unroll
      for (int ks = 0; ks < 4; ++ks) {
        v8h l0v = *(const v8h*)(b0 + ks * 32);
        v8h h0v = *(const v8h*)(b0 + ks * 32 + 16);
        v8h l1v = *(const v8h*)(b1 + ks * 32);
        v8h h1v = *(const v8h*)(b1 + ks * 32 + 16);
        v16h bf0 = __builtin_shufflevector(l0v, h0v, 0, 1, 2, 3, 4, 5, 6, 7,
                                           8, 9, 10, 11, 12, 13, 14, 15);
        v16h bf1 = __builtin_shufflevector(l1v, h1v, 0, 1, 2, 3, 4, 5, 6, 7,
                                           8, 9, 10, 11, 12, 13, 14, 15);
        c0 = __builtin_amdgcn_wmma_f32_16x16x32_f16(
            false, a[ks], false, bf0, (short)0, c0, false, false);
        c1 = __builtin_amdgcn_wmma_f32_16x16x32_f16(
            false, a[ks], false, bf1, (short)0, c1, false, false);
      }
      const int n0 = p * KHALF + nt * 16 + lm;  // lane's column this tile pair
#pragma unroll
      for (int j = 0; j < 8; ++j) {             // row M = j + half*8
        if (c0[j] > mv[j]) { mv[j] = c0[j]; mi[j] = n0; }
        if (c1[j] > mv[j]) { mv[j] = c1[j]; mi[j] = n0 + 16; }
      }
    }
  }

  // ---- argmax across the 16 lanes sharing each row group (tie -> min idx) --
#pragma unroll
  for (int j = 0; j < 8; ++j) {
    float v = mv[j];
    int   i = mi[j];
#pragma unroll
    for (int off = 8; off >= 1; off >>= 1) {
      const float ov = __shfl_xor(v, off, 32);
      const int   oi = __shfl_xor(i, off, 32);
      if (ov > v || (ov == v && oi < i)) { v = ov; i = oi; }
    }
    mv[j] = v;
    mi[j] = i;
  }
  if (lm == 0) {
#pragma unroll
    for (int j = 0; j < 8; ++j) {
      const int ll = wid * 16 + half * 8 + j;
      idx_lds[ll] = mi[j];
      idxout[((size_t)b * H_ + h) * L_ + l0 + ll] = (float)mi[j];
    }
  }
  __syncthreads();   // idx ready; K region dead -> reuse as o_lds

  // ---- gather v[h, idx[ll], :] into LDS (2 threads per row) ----
  {
    const int ll = t >> 1, hf = t & 1;
    const int n = idx_lds[ll];
    const float4* src =
        (const float4*)(vf + ((size_t)h * NCODES_ + n) * D_ + hf * 64);
    float* dst = o_lds + ll * OROW + hf * 64;
#pragma unroll
    for (int i = 0; i < 16; ++i) {
      const float4 u = src[i];
      dst[i * 4 + 0] = u.x; dst[i * 4 + 1] = u.y;
      dst[i * 4 + 2] = u.z; dst[i * 4 + 3] = u.w;
    }
  }
  __syncthreads();

  // ---- transposed write-out: out[b, h*128+d, l0+ll], coalesced along l ----
  {
    const int ll = t & 127;
    const int dg = t >> 7;   // 0 or 1 -> d in [0,64) or [64,128)
    float* ob =
        out + ((size_t)b * DIM_ + (size_t)h * D_ + (size_t)dg * 64) * L_ + l0 + ll;
    const float* srow = o_lds + ll * OROW + dg * 64;
#pragma unroll
    for (int j = 0; j < 64; ++j)
      ob[(size_t)j * L_] = srow[j];   // OROW=129 -> conflict-free LDS reads
  }
}

// ---------------------------------------------------------------------------
extern "C" void kernel_launch(void* const* d_in, const int* in_sizes, int n_in,
                              void* d_out, int out_size, void* d_ws,
                              size_t ws_size, hipStream_t stream) {
  const float* x     = (const float*)d_in[0];
  const float* codes = (const float*)d_in[1];
  const float* Wk    = (const float*)d_in[2];
  const float* Wv    = (const float*)d_in[3];

  float* out    = (float*)d_out;
  float* idxout = out + (size_t)B_ * DIM_ * L_;   // idx section (as f32 values)

  f16*   kfw = (f16*)d_ws;                                            // 1 MB
  float* vfw = (float*)((char*)d_ws + (size_t)H_ * NCODES_ * D_ * 2); // 2 MB

  build_kv_kernel<<<NCODES_, 256, 0, stream>>>(codes, Wk, Wv, kfw, vfw);

  static_assert(SMEM_BYTES <= 320 * 1024, "LDS budget");
  hipFuncSetAttribute((const void*)hard_attn_kernel,
                      hipFuncAttributeMaxDynamicSharedMemorySize, SMEM_BYTES);
  dim3 grid(L_ / LTILE, H_, B_);
  hard_attn_kernel<<<grid, 256, SMEM_BYTES, stream>>>(x, kfw, vfw, out, idxout);
}
